// CrossAttention_44040594653424
// MI455X (gfx1250) — matrix operand order repaired and checked by reading, not compiled, not run on hardware
//
#include <hip/hip_runtime.h>
#include <hip/hip_bf16.h>

// ---------------------------------------------------------------------------
// Problem constants: B=4, C=256, Cr=32, D=H=W=16 -> N=4096
// ---------------------------------------------------------------------------
#define BATCH 4
#define CCH   256
#define CR    32
#define NSP   4096
#define EPSN  1e-5f

typedef __attribute__((ext_vector_type(16))) __bf16 v16bf;
typedef __attribute__((ext_vector_type(8)))  __bf16 v8bf;
typedef __attribute__((ext_vector_type(8)))  float  v8f;

__device__ __forceinline__ v16bf cat8(v8bf lo, v8bf hi) {
    return __builtin_shufflevector(lo, hi, 0,1,2,3,4,5,6,7,8,9,10,11,12,13,14,15);
}

// ---------------------------------------------------------------------------
// Kernel 1: channel matmul (1x1 conv3d) fp32 -> bf16, two output layouts.
//   transposeOut=1: out[b][n][Cout]  (for Q^T / K A/B fragments)
//   transposeOut=0: out[b][co][n]    (for V B fragments)
// ---------------------------------------------------------------------------
__global__ __launch_bounds__(256) void chan_matmul_bf16(
    const float* __restrict__ x,   // [B][C][N]
    const float* __restrict__ w,   // [Cout][C]
    __bf16* __restrict__ out,
    int Cout, int transposeOut)
{
    int tid = blockIdx.x * blockDim.x + threadIdx.x;
    int total = BATCH * Cout * NSP;
    if (tid >= total) return;
    int n  = tid & (NSP - 1);
    int co = (tid >> 12) % Cout;
    int b  = tid / (Cout << 12);
    const float* xp  = x + ((size_t)b * CCH) * NSP + n;
    const float* wpp = w + (size_t)co * CCH;
    float acc = 0.f;
    #pragma unroll 8
    for (int c = 0; c < CCH; ++c)
        acc += wpp[c] * xp[(size_t)c * NSP];
    size_t oi = transposeOut ? ((size_t)(b * NSP + n) * Cout + co)
                             : ((size_t)(b * Cout + co) * NSP + n);
    out[oi] = (__bf16)acc;
}

// ---------------------------------------------------------------------------
// Kernel 2: transpose wp [256][256] -> wpT [c][co] for coalesced projection
// ---------------------------------------------------------------------------
__global__ __launch_bounds__(256) void transpose256(
    const float* __restrict__ w, float* __restrict__ wt)
{
    int tid = blockIdx.x * blockDim.x + threadIdx.x;
    int c = tid & 255, co = tid >> 8;
    wt[c * 256 + co] = w[co * 256 + c];
}

// ---------------------------------------------------------------------------
// Kernel 3: fused attention (flash-style, exact two-pass softmax) with WMMA.
// One workgroup = 8 wave32s handles (b, i-tile of 16 query rows).
//   Pass 1: S = Q^T K via v_wmma_f32_16x16x32_bf16; PER-LANE running max only
//           (lane reduction deferred to a single post-loop shuffle tree).
//   Pass 2: recompute S, P~ = exp(S - max) -> bf16 into LDS; PER-LANE sums,
//           one post-loop shuffle tree.
//   Phase B: each wave owns 32 channels; O = P~ x V^T via WMMA over all j,
//            scale rows by 1/rowsum in epilogue. Writes attnT[b][n][c].
// ---------------------------------------------------------------------------
__global__ __launch_bounds__(256) void attn_kernel(
    const __bf16* __restrict__ qT,   // [B][N][CR]
    const __bf16* __restrict__ kT,   // [B][N][CR]
    const __bf16* __restrict__ vbf,  // [B][C][N]
    float* __restrict__ attnT)       // [B][N][C]
{
    extern __shared__ char smem[];
    __bf16* P    = (__bf16*)smem;                         // [16][4096] bf16 = 128KB
    float*  red  = (float*)(smem + 16 * NSP * 2);         // [8][16]
    float*  gmax = (float*)(smem + 16 * NSP * 2 + 512);   // [16]
    float*  ginv = (float*)(smem + 16 * NSP * 2 + 576);   // [16]

    const int tid  = threadIdx.x;
    const int lane = tid & 31;
    const int wave = tid >> 5;
    const int half = (lane >> 4) & 1;
    const int l16  = lane & 15;

    const int b  = blockIdx.x >> 8;        // 4 * 256 blocks
    const int it = blockIdx.x & 255;
    const int i0 = it << 4;

    // scale = Cr^-0.5 ; exp(x*scale) = exp2(x*scale*log2(e))
    const float K2 = 0.17677669529663687f * 1.4426950408889634f;

    // A-fragment: Q^T tile [M=16 rows i][K=32 channels], loop-invariant.
    // lane<16: M=l16, K={0..7,16..23}; lane>=16: M=l16, K={8..15,24..31}
    v16bf aQ;
    {
        const __bf16* p = qT + ((size_t)(b * NSP + i0 + l16) << 5) + (half ? 8 : 0);
        v8bf lo = *(const v8bf*)p;
        v8bf hi = *(const v8bf*)(p + 16);
        aQ = cat8(lo, hi);
    }

    v8f z8 = {0.f, 0.f, 0.f, 0.f, 0.f, 0.f, 0.f, 0.f};

    // ---------------- pass 1: per-lane running max, reduce once ----------------
    float rm[8];
    #pragma unroll
    for (int r = 0; r < 8; ++r) rm[r] = -3.0e38f;

    for (int t = wave * 32; t < wave * 32 + 32; ++t) {
        int j0 = t << 4;
        // B-fragment: K tile [K=32 channels][N=16 cols j]
        const __bf16* p = kT + ((size_t)(b * NSP + j0 + l16) << 5) + (half ? 8 : 0);
        v16bf bK = cat8(*(const v8bf*)p, *(const v8bf*)(p + 16));
        v8f s = __builtin_amdgcn_wmma_f32_16x16x32_bf16(
                    false, aQ, false, bK, (short)0, z8, false, false);
        #pragma unroll
        for (int r = 0; r < 8; ++r) rm[r] = fmaxf(rm[r], s[r]);
    }
    // one shuffle tree per row register (cross-lane, within 16-lane half)
    #pragma unroll
    for (int r = 0; r < 8; ++r) {
        float x = rm[r];
        x = fmaxf(x, __shfl_xor(x, 1));
        x = fmaxf(x, __shfl_xor(x, 2));
        x = fmaxf(x, __shfl_xor(x, 4));
        x = fmaxf(x, __shfl_xor(x, 8));
        rm[r] = x;
    }
    if (lane == 0)  { for (int r = 0; r < 8; ++r) red[wave * 16 + r]     = rm[r]; }
    if (lane == 16) { for (int r = 0; r < 8; ++r) red[wave * 16 + 8 + r] = rm[r]; }
    __syncthreads();
    if (tid < 16) {
        float m = -3.0e38f;
        for (int w = 0; w < 8; ++w) m = fmaxf(m, red[w * 16 + tid]);
        gmax[tid] = m;
    }
    __syncthreads();
    float gm[8];
    #pragma unroll
    for (int r = 0; r < 8; ++r) gm[r] = gmax[half * 8 + r];

    // ---------------- pass 2: exp, store P~, per-lane sums ----------------
    float rs[8];
    #pragma unroll
    for (int r = 0; r < 8; ++r) rs[r] = 0.f;

    for (int t = wave * 32; t < wave * 32 + 32; ++t) {
        int j0 = t << 4;
        const __bf16* p = kT + ((size_t)(b * NSP + j0 + l16) << 5) + (half ? 8 : 0);
        v16bf bK = cat8(*(const v8bf*)p, *(const v8bf*)(p + 16));
        v8f s = __builtin_amdgcn_wmma_f32_16x16x32_bf16(
                    false, aQ, false, bK, (short)0, z8, false, false);
        #pragma unroll
        for (int r = 0; r < 8; ++r) {
            float e = exp2f((s[r] - gm[r]) * K2);
            int row = r + half * 8;                 // C/D layout: VGPR r -> row r / r+8
            P[row * NSP + j0 + l16] = (__bf16)e;    // unnormalized probability
            rs[r] += e;                             // per-lane partial, no shuffle here
        }
    }
    #pragma unroll
    for (int r = 0; r < 8; ++r) {
        float x = rs[r];
        x += __shfl_xor(x, 1);
        x += __shfl_xor(x, 2);
        x += __shfl_xor(x, 4);
        x += __shfl_xor(x, 8);
        rs[r] = x;
    }
    __syncthreads();  // P visible; red reusable
    if (lane == 0)  { for (int r = 0; r < 8; ++r) red[wave * 16 + r]     = rs[r]; }
    if (lane == 16) { for (int r = 0; r < 8; ++r) red[wave * 16 + 8 + r] = rs[r]; }
    __syncthreads();
    if (tid < 16) {
        float s = 0.f;
        for (int w = 0; w < 8; ++w) s += red[w * 16 + tid];
        ginv[tid] = 1.0f / s;
    }
    __syncthreads();

    // ---------------- phase B: O = P~ x V^T over all j ----------------
    const int c0a = wave * 32;        // this wave's channel tiles
    const int c0b = wave * 32 + 16;
    v8f acc0 = z8, acc1 = z8;

    #pragma unroll 2
    for (int jb = 0; jb < NSP; jb += 32) {
        // A-fragment: P tile [M=16 rows i][K=32 cols j] from LDS
        v16bf aP;
        {
            const __bf16* p = P + l16 * NSP + jb + half * 16;
            aP = cat8(*(const v8bf*)p, *(const v8bf*)(p + 8));
        }
        // B-fragments: V^T tile [K=32 j][N=16 channels]; element [j][c] = v[c][j]
        const __bf16* pv0 = vbf + ((size_t)(b * CCH + c0a + l16) << 12) + jb + half * 16;
        v16bf bV0 = cat8(*(const v8bf*)pv0, *(const v8bf*)(pv0 + 8));
        const __bf16* pv1 = vbf + ((size_t)(b * CCH + c0b + l16) << 12) + jb + half * 16;
        v16bf bV1 = cat8(*(const v8bf*)pv1, *(const v8bf*)(pv1 + 8));

        acc0 = __builtin_amdgcn_wmma_f32_16x16x32_bf16(
                   false, aP, false, bV0, (short)0, acc0, false, false);
        acc1 = __builtin_amdgcn_wmma_f32_16x16x32_bf16(
                   false, aP, false, bV1, (short)0, acc1, false, false);
    }

    #pragma unroll
    for (int r = 0; r < 8; ++r) {
        int row = r + half * 8;
        float gi = ginv[row];
        int n = i0 + row;
        attnT[((size_t)(b * NSP + n) << 8) + c0a + l16] = acc0[r] * gi;
        attnT[((size_t)(b * NSP + n) << 8) + c0b + l16] = acc1[r] * gi;
    }
}

// ---------------------------------------------------------------------------
// Kernel 4: output projection  out[b][co][n] = sum_c wpT[c][co]*attnT[b][n][c]
// One block per (b,n); 256 threads = co. attnT row staged in LDS (broadcast).
// ---------------------------------------------------------------------------
__global__ __launch_bounds__(256) void proj_kernel(
    const float* __restrict__ attnT,  // [B][N][C]
    const float* __restrict__ wpT,    // [C][C]  (wpT[c][co])
    float* __restrict__ out)          // [B][C][N]
{
    __shared__ float xrow[CCH];
    int bn = blockIdx.x;              // b*4096 + n
    int t  = threadIdx.x;
    xrow[t] = attnT[(size_t)bn * CCH + t];
    __syncthreads();
    int b = bn >> 12, n = bn & (NSP - 1);
    float acc = 0.f;
    #pragma unroll 8
    for (int c = 0; c < CCH; ++c)
        acc += xrow[c] * wpT[c * CCH + t];
    out[((size_t)(b * CCH + t)) * NSP + n] = acc;
}

// ---------------------------------------------------------------------------
// Kernel 5: InstanceNorm3d (biased var, no affine) + residual, in-place on out.
// One block per (b,c); fixed-order tree reduction (deterministic, no atomics).
// ---------------------------------------------------------------------------
__global__ __launch_bounds__(256) void norm_kernel(
    float* __restrict__ out, const float* __restrict__ x1)
{
    __shared__ float sb[256], qb[256];
    int bc = blockIdx.x;
    int t  = threadIdx.x;
    size_t base = (size_t)bc * NSP;
    float xv[16];
    float s = 0.f, q = 0.f;
    #pragma unroll
    for (int k = 0; k < 16; ++k) {
        xv[k] = out[base + t + k * 256];
        s += xv[k];
        q += xv[k] * xv[k];
    }
    sb[t] = s; qb[t] = q;
    __syncthreads();
    for (int off = 128; off > 0; off >>= 1) {
        if (t < off) { sb[t] += sb[t + off]; qb[t] += qb[t + off]; }
        __syncthreads();
    }
    float mean = sb[0] * (1.f / NSP);
    float var  = qb[0] * (1.f / NSP) - mean * mean;
    float rstd = rsqrtf(var + EPSN);
    #pragma unroll
    for (int k = 0; k < 16; ++k) {
        size_t i = base + t + k * 256;
        out[i] = (xv[k] - mean) * rstd + x1[i];
    }
}

// ---------------------------------------------------------------------------
// Host launcher. Workspace layout (bytes):
//   qT    bf16 [4][4096][32]   @ 0         (1,048,576)
//   kT    bf16 [4][4096][32]   @ 1,048,576 (1,048,576)
//   vbf   bf16 [4][256][4096]  @ 2,097,152 (8,388,608)
//   attnT f32  [4][4096][256]  @ 10,485,760 (16,777,216)
//   wpT   f32  [256][256]      @ 27,262,976 (262,144)
// total ~27.5 MB
// ---------------------------------------------------------------------------
extern "C" void kernel_launch(void* const* d_in, const int* in_sizes, int n_in,
                              void* d_out, int out_size, void* d_ws, size_t ws_size,
                              hipStream_t stream) {
    (void)in_sizes; (void)n_in; (void)out_size; (void)ws_size;
    const float* x1 = (const float*)d_in[0];
    const float* x2 = (const float*)d_in[1];
    const float* wq = (const float*)d_in[2];
    const float* wk = (const float*)d_in[3];
    const float* wv = (const float*)d_in[4];
    const float* wp = (const float*)d_in[5];
    float* out = (float*)d_out;

    unsigned char* ws = (unsigned char*)d_ws;
    __bf16* qT    = (__bf16*)(ws);
    __bf16* kT    = (__bf16*)(ws + 1048576);
    __bf16* vbf   = (__bf16*)(ws + 2097152);
    float*  attnT = (float*)(ws + 10485760);
    float*  wpT   = (float*)(ws + 27262976);

    // QKV projections (fp32 accumulate, bf16 outputs in WMMA-friendly layouts)
    chan_matmul_bf16<<<(BATCH * CR  * NSP) / 256, 256, 0, stream>>>(x1, wq, qT,  CR,  1);
    chan_matmul_bf16<<<(BATCH * CR  * NSP) / 256, 256, 0, stream>>>(x2, wk, kT,  CR,  1);
    chan_matmul_bf16<<<(BATCH * CCH * NSP) / 256, 256, 0, stream>>>(x2, wv, vbf, CCH, 0);
    transpose256<<<(CCH * CCH) / 256, 256, 0, stream>>>(wp, wpT);

    // Fused flash attention: 4 batches x 256 i-tiles, 132KB dynamic LDS
    size_t shmem = (size_t)16 * NSP * 2 + 512 + 64 + 64;
    attn_kernel<<<BATCH * (NSP / 16), 256, shmem, stream>>>(qT, kT, vbf, attnT);

    // Output projection, then InstanceNorm + residual (in place on d_out)
    proj_kernel<<<BATCH * NSP, 256, 0, stream>>>(attnT, wpT, out);
    norm_kernel<<<BATCH * CCH, 256, 0, stream>>>(out, x1);
}
